// EdgeModel_72567767433246
// MI455X (gfx1250) — matrix operand order repaired and checked by reading, not compiled, not run on hardware
//
#include <hip/hip_runtime.h>
#include <stdint.h>

// ---------------- CDNA5 (gfx1250) types ----------------
typedef __attribute__((ext_vector_type(16))) __bf16 v16bf;
typedef __attribute__((ext_vector_type(8)))  float  v8f;
typedef __attribute__((ext_vector_type(4)))  unsigned int u32x4;
typedef __attribute__((ext_vector_type(2)))  unsigned int u32x2;
typedef __attribute__((ext_vector_type(4)))  float  f32x4;

union FragAB {
    v16bf v;
    u32x4 q[2];
};

__device__ __forceinline__ unsigned short f32_to_bf16_bits(float f) {
    union { float f; uint32_t u; } v; v.f = f;
    // round-to-nearest-even
    uint32_t r = v.u + 0x7FFFu + ((v.u >> 16) & 1u);
    return (unsigned short)(r >> 16);
}

// Problem dims
#define D_FEAT   256
#define IN_DIM   1024   // 4*D
#define H_DIM    512
#define O_DIM    256
#define MT       64     // edges per block
// LDS row strides (in bf16 elements), padded by 8 (16B) to rotate banks
#define ASTR     1032   // 1024 + 8
#define HSTR     520    // 512  + 8
#define SMEM_BYTES ((MT*ASTR + MT*HSTR) * 2)

// ---------------- prep: fp32 [K][N] -> bf16 transposed [N][K] ----------------
__global__ void transpose_to_bf16(const float* __restrict__ in,
                                  unsigned short* __restrict__ outw,
                                  int K, int N) {
    int idx = blockIdx.x * 256 + threadIdx.x;
    if (idx >= K * N) return;
    int n = idx / K;           // output row (N-major)
    int k = idx - n * K;       // contiguous K within row
    outw[idx] = f32_to_bf16_bits(in[(size_t)k * N + n]);
}

// ---------------- fused edge-MLP ----------------
__launch_bounds__(256, 1)
__global__ void edge_mlp_kernel(const float* __restrict__ srcf,
                                const float* __restrict__ dstf,
                                const float* __restrict__ eaf,
                                const float* __restrict__ uf,
                                const int*   __restrict__ batch,
                                const unsigned short* __restrict__ w1t, // [H][IN] bf16
                                const float* __restrict__ b1,
                                const unsigned short* __restrict__ w2t, // [O][H] bf16
                                const float* __restrict__ b2,
                                float* __restrict__ out,
                                int Etot) {
    extern __shared__ unsigned short smem[];
    unsigned short* Ash = smem;                 // [MT][ASTR] bf16 concat inputs
    unsigned short* Hsh = smem + MT * ASTR;     // [MT][HSTR] bf16 hidden acts

    const int tid  = threadIdx.x;
    const int base = blockIdx.x * MT;
    const int lane = tid & 31;
    const int wave = tid >> 5;
    const int l16  = lane & 15;
    const int hi   = lane >> 4;   // 0 for lanes 0-15, 1 for lanes 16-31

    // ---- Phase 0: stage A tile = [src|dest|edge_attr|u[batch]] as bf16 in LDS ----
    {
        const int col = tid << 2;        // each thread owns one float4 column chunk
        const int seg = tid >> 6;        // 0..3
        const int cs  = col & 255;       // column within segment
        for (int row = 0; row < MT; ++row) {
            int e = base + row;
            if (e >= Etot) e = Etot - 1;
            const float* p;
            if (seg == 0)      p = srcf + (size_t)e * D_FEAT + cs;
            else if (seg == 1) p = dstf + (size_t)e * D_FEAT + cs;
            else if (seg == 2) p = eaf  + (size_t)e * D_FEAT + cs;
            else               p = uf   + (size_t)batch[e] * D_FEAT + cs;
            f32x4 d = *(const f32x4*)p;
            u32x2 packed;
            packed.x = (unsigned)f32_to_bf16_bits(d[0]) |
                       ((unsigned)f32_to_bf16_bits(d[1]) << 16);
            packed.y = (unsigned)f32_to_bf16_bits(d[2]) |
                       ((unsigned)f32_to_bf16_bits(d[3]) << 16);
            *(u32x2*)(Ash + row * ASTR + col) = packed;
        }
    }
    __syncthreads();

    // ---- Phase 1: GEMM1  h = relu(A[64x1024] @ W1[1024x512] + b1) -> Hsh bf16 ----
    // Each wave: 2 passes, each pass holds 2 N-tiles x 4 M-tiles of accumulators,
    // so every A fragment read from LDS feeds 2 WMMAs.
    for (int p = 0; p < 2; ++p) {
        const int nA = (wave + p * 8) * 16 + l16;          // N-strip 0
        const int nB = (wave + p * 8 + 16) * 16 + l16;     // N-strip 1
        v8f acc[2][4] = {};
        const unsigned short* wrow0 = w1t + (size_t)nA * IN_DIM + hi * 16;
        const unsigned short* wrow1 = w1t + (size_t)nB * IN_DIM + hi * 16;
        const int aoff = hi * 16;
        for (int k0 = 0; k0 < IN_DIM; k0 += 32) {
            FragAB bf0, bf1;
            { const u32x4* bp = (const u32x4*)(wrow0 + k0); bf0.q[0] = bp[0]; bf0.q[1] = bp[1]; }
            { const u32x4* bp = (const u32x4*)(wrow1 + k0); bf1.q[0] = bp[0]; bf1.q[1] = bp[1]; }

            FragAB a[4];
            #pragma unroll
            for (int m = 0; m < 4; ++m) {
                const unsigned short* pA = Ash + k0 + aoff + (size_t)(m * 16 + l16) * ASTR;
                a[m].q[0] = *(const u32x4*)pA;
                a[m].q[1] = *(const u32x4*)(pA + 8);
            }
            #pragma unroll
            for (int m = 0; m < 4; ++m) {
                acc[0][m] = __builtin_amdgcn_wmma_f32_16x16x32_bf16(false, a[m].v, false, bf0.v,
                                                                    (short)0, acc[0][m], false, false);
                acc[1][m] = __builtin_amdgcn_wmma_f32_16x16x32_bf16(false, a[m].v, false, bf1.v,
                                                                    (short)0, acc[1][m], false, false);
            }
        }
        const float biasA = b1[nA];
        const float biasB = b1[nB];
        #pragma unroll
        for (int r = 0; r < 8; ++r) {
            const int rb = r + 8 * hi;   // C layout: M = r + 8*(lane>=16)
            #pragma unroll
            for (int m = 0; m < 4; ++m) {
                float vA = acc[0][m][r] + biasA; vA = vA > 0.f ? vA : 0.f;
                float vB = acc[1][m][r] + biasB; vB = vB > 0.f ? vB : 0.f;
                Hsh[(size_t)(m * 16 + rb) * HSTR + nA] = f32_to_bf16_bits(vA);
                Hsh[(size_t)(m * 16 + rb) * HSTR + nB] = f32_to_bf16_bits(vB);
            }
        }
    }
    __syncthreads();

    // ---- Phase 2: GEMM2  out = h[64x512] @ W2[512x256] + b2 (single pass/wave) ----
    {
        const int nA = wave * 16 + l16;           // 0..127
        const int nB = (wave + 8) * 16 + l16;     // 128..255
        v8f acc[2][4] = {};
        const unsigned short* wrow0 = w2t + (size_t)nA * H_DIM + hi * 16;
        const unsigned short* wrow1 = w2t + (size_t)nB * H_DIM + hi * 16;
        const int aoff = hi * 16;
        for (int k0 = 0; k0 < H_DIM; k0 += 32) {
            FragAB bf0, bf1;
            { const u32x4* bp = (const u32x4*)(wrow0 + k0); bf0.q[0] = bp[0]; bf0.q[1] = bp[1]; }
            { const u32x4* bp = (const u32x4*)(wrow1 + k0); bf1.q[0] = bp[0]; bf1.q[1] = bp[1]; }

            FragAB a[4];
            #pragma unroll
            for (int m = 0; m < 4; ++m) {
                const unsigned short* pA = Hsh + k0 + aoff + (size_t)(m * 16 + l16) * HSTR;
                a[m].q[0] = *(const u32x4*)pA;
                a[m].q[1] = *(const u32x4*)(pA + 8);
            }
            #pragma unroll
            for (int m = 0; m < 4; ++m) {
                acc[0][m] = __builtin_amdgcn_wmma_f32_16x16x32_bf16(false, a[m].v, false, bf0.v,
                                                                    (short)0, acc[0][m], false, false);
                acc[1][m] = __builtin_amdgcn_wmma_f32_16x16x32_bf16(false, a[m].v, false, bf1.v,
                                                                    (short)0, acc[1][m], false, false);
            }
        }
        const float biasA = b2[nA];
        const float biasB = b2[nB];
        #pragma unroll
        for (int r = 0; r < 8; ++r) {
            const int rb = r + 8 * hi;
            #pragma unroll
            for (int m = 0; m < 4; ++m) {
                const int e = base + m * 16 + rb;
                if (e < Etot) {
                    out[(size_t)e * O_DIM + nA] = acc[0][m][r] + biasA;
                    out[(size_t)e * O_DIM + nB] = acc[1][m][r] + biasB;
                }
            }
        }
    }
}

// ---------------- host-side launch ----------------
extern "C" void kernel_launch(void* const* d_in, const int* in_sizes, int n_in,
                              void* d_out, int out_size, void* d_ws, size_t ws_size,
                              hipStream_t stream) {
    const float* srcf  = (const float*)d_in[0];
    const float* dstf  = (const float*)d_in[1];
    const float* eaf   = (const float*)d_in[2];
    const float* uf    = (const float*)d_in[3];
    const int*   batch = (const int*)d_in[4];
    const float* W1    = (const float*)d_in[5];
    const float* b1    = (const float*)d_in[6];
    const float* W2    = (const float*)d_in[7];
    const float* b2    = (const float*)d_in[8];
    float* out = (float*)d_out;

    const int Etot = in_sizes[0] / D_FEAT;

    unsigned short* w1t = (unsigned short*)d_ws;                 // [512][1024] bf16
    unsigned short* w2t = w1t + (size_t)H_DIM * IN_DIM;          // [256][512]  bf16

    // Convert + transpose weights (deterministic, every call)
    transpose_to_bf16<<<(IN_DIM * H_DIM + 255) / 256, 256, 0, stream>>>(W1, w1t, IN_DIM, H_DIM);
    transpose_to_bf16<<<(H_DIM * O_DIM + 255) / 256, 256, 0, stream>>>(W2, w2t, H_DIM, O_DIM);

    (void)hipFuncSetAttribute((const void*)edge_mlp_kernel,
                              hipFuncAttributeMaxDynamicSharedMemorySize, SMEM_BYTES);

    const int nblocks = (Etot + MT - 1) / MT;   // 3125 for E=200000
    edge_mlp_kernel<<<nblocks, 256, SMEM_BYTES, stream>>>(
        srcf, dstf, eaf, uf, batch, w1t, b1, w2t, b2, out, Etot);
}
